// MLPOneHotEmbeddingModule_78752520339916
// MI455X (gfx1250) — matrix operand order repaired and checked by reading, not compiled, not run on hardware
//
#include <hip/hip_runtime.h>
#include <hip/hip_bf16.h>
#include <cstdint>

typedef __attribute__((ext_vector_type(16))) _Float16 v16h;
typedef __attribute__((ext_vector_type(8)))  _Float16 v8h;
typedef __attribute__((ext_vector_type(4)))  _Float16 v4h;
typedef __attribute__((ext_vector_type(8)))  float    v8f;
typedef int v4i_vs __attribute__((vector_size(4 * sizeof(int))));

constexpr int HID   = 1024;   // hidden dim (K for both GEMMs, N for GEMM1)
constexpr int OUTD  = 256;    // output dim
constexpr int BATCH = 8192;
constexpr float SLOPE = 0.2f;
constexpr float EPSV  = 1e-5f;
constexpr int ASTRIDE = HID + 8;   // padded f16 row stride in LDS (1032 halves = 2064 B, 16B aligned)
constexpr int HSTRIDE = HID + 8;   // padded f32 row stride in LDS

union F16Frag { v16h v; v8h h[2]; };

__device__ __forceinline__ float lrelu(float v) { return v > 0.0f ? v : SLOPE * v; }

// 16-byte global -> LDS staging, async on gfx1250 when the builtin exists.
__device__ __forceinline__ void stage16B(const _Float16* g, _Float16* l) {
#if __has_builtin(__builtin_amdgcn_global_load_async_to_lds_b128)
    __builtin_amdgcn_global_load_async_to_lds_b128(
        (v4i_vs*)g, (v4i_vs*)l, 0, 0);
#else
    *(v8h*)l = *(const v8h*)g;
#endif
}

__device__ __forceinline__ void stage_join() {
#if __has_builtin(__builtin_amdgcn_global_load_async_to_lds_b128)
 #if __has_builtin(__builtin_amdgcn_s_wait_asynccnt)
    __builtin_amdgcn_s_wait_asynccnt(0);
 #else
    asm volatile("s_wait_asynccnt 0" ::: "memory");
 #endif
#endif
}

// ---------------------------------------------------------------------------
// Weight conversion: W2 [K][N] f32 -> W2T [N][K] f16   (K-contiguous for B frags)
// ---------------------------------------------------------------------------
__global__ void k_convW2(const float* __restrict__ W2, _Float16* __restrict__ W2T) {
    int i = blockIdx.x * 256 + threadIdx.x;          // [0, HID*HID)
    int k = i & (HID - 1);
    int n = i >> 10;
    W2T[(size_t)n * HID + k] = (_Float16)W2[(size_t)k * HID + n];
}

__global__ void k_convW3(const float* __restrict__ W3, _Float16* __restrict__ W3T) {
    int i = blockIdx.x * 256 + threadIdx.x;          // [0, OUTD*HID)
    int k = i & (HID - 1);
    int o = i >> 10;
    W3T[(size_t)o * HID + k] = (_Float16)W3[(size_t)k * OUTD + o];
}

// ---------------------------------------------------------------------------
// Embedding gather + b1 + leaky_relu -> f16 activations A [BATCH][HID]
// ---------------------------------------------------------------------------
__global__ void k_embed(const int* __restrict__ x, const float* __restrict__ W1,
                        const float* __restrict__ b1, _Float16* __restrict__ A) {
    int row = blockIdx.x;
    int t   = threadIdx.x;
    size_t base = (size_t)x[row] * HID;
    float4 w = ((const float4*)(W1 + base))[t];
    float4 b = ((const float4*)b1)[t];
    v4h o;
    o[0] = (_Float16)lrelu(w.x + b.x);
    o[1] = (_Float16)lrelu(w.y + b.y);
    o[2] = (_Float16)lrelu(w.z + b.z);
    o[3] = (_Float16)lrelu(w.w + b.w);
    ((v4h*)(A + (size_t)row * HID))[t] = o;
}

// ---------------------------------------------------------------------------
// GEMM1 (A[16 x 1024] @ W2[1024 x 1024]) + b2 + LayerNorm + leaky_relu -> f16
// block = 256 threads = 8 waves; block owns 16 rows x full 1024 cols.
// A tile staged once in LDS (async); each wave owns 8 accumulator tiles.
// ---------------------------------------------------------------------------
__global__ void __launch_bounds__(256, 1)
k_gemm1_ln(const _Float16* __restrict__ A,
           const _Float16* __restrict__ W2T,
           const float* __restrict__ b2,
           const float* __restrict__ g2,
           const float* __restrict__ beta2,
           _Float16* __restrict__ Hout) {
    extern __shared__ char smemraw[];
    _Float16* As = (_Float16*)smemraw;                         // 16 x ASTRIDE f16
    float* Hs   = (float*)(smemraw + 16 * ASTRIDE * 2);        // 16 x HSTRIDE f32
    float* red  = Hs + 16 * HSTRIDE;                           // 16 x 16
    float* red2 = red + 256;                                   // 16 x 16
    float* muS  = red2 + 256;                                  // 16
    float* invS = muS + 16;                                    // 16

    const int tid   = threadIdx.x;
    const int lane  = tid & 31;
    const int wave  = tid >> 5;
    const int mBase = blockIdx.x * 16;
    const int l15   = lane & 15;
    const int g     = lane >> 4;

    // ---- Stage the 16x1024 f16 A tile into LDS (2048 x 16B chunks) ----
    for (int c = tid; c < 16 * (HID / 8); c += 256) {          // 2048 chunks
        int r  = c >> 7;                                       // 128 chunks per row
        int co = (c & 127) * 8;
        stage16B(A + (size_t)(mBase + r) * HID + co, As + r * ASTRIDE + co);
    }
    stage_join();
    __syncthreads();

    // B fragment base: col n = wave*16 + (lane&15); halves at K = g*16 .. +15
    const _Float16* bPtr = W2T + (size_t)(wave * 16 + l15) * HID + g * 16;
    // A fragment base in LDS: row = lane&15; halves at K = g*16 .. +15
    const _Float16* aLds = As + l15 * ASTRIDE + g * 16;

    v8f acc[8] = {};
#pragma unroll 1
    for (int kb = 0; kb < HID / 32; ++kb) {
        F16Frag af;
        af.h[0] = *(const v8h*)(aLds + kb * 32);
        af.h[1] = *(const v8h*)(aLds + kb * 32 + 8);
        // Batch all 8 B-fragment loads so waits stagger instead of draining.
        F16Frag bf[8];
#pragma unroll
        for (int nc = 0; nc < 8; ++nc) {
            const _Float16* bp = bPtr + (size_t)nc * (128 * HID) + kb * 32;
            bf[nc].h[0] = *(const v8h*)(bp);
            bf[nc].h[1] = *(const v8h*)(bp + 8);
        }
        if (kb + 4 < HID / 32) {
#pragma unroll
            for (int nc = 0; nc < 8; ++nc)
                __builtin_prefetch(bPtr + (size_t)nc * (128 * HID) + (kb + 4) * 32, 0, 3);
        }
#pragma unroll
        for (int nc = 0; nc < 8; ++nc) {
            acc[nc] = __builtin_amdgcn_wmma_f32_16x16x32_f16(
                false, af.v, false, bf[nc].v, (short)0, acc[nc], false, false);
        }
    }

    // Epilogue: + b2, stash pre-LN f32 rows in LDS.
    // C layout: VGPR v -> row = v + g*8 ; col = tile_n + (lane&15)
#pragma unroll
    for (int nc = 0; nc < 8; ++nc) {
        int col = nc * 128 + wave * 16 + l15;
        float bias = b2[col];
#pragma unroll
        for (int v = 0; v < 8; ++v) {
            int row = v + g * 8;
            Hs[row * HSTRIDE + col] = acc[nc][v] + bias;
        }
    }
    __syncthreads();

    // LayerNorm: two-stage reduction over each 1024-wide row.
    {
        int r = tid & 15, ch = tid >> 4;
        float s = 0.0f, s2 = 0.0f;
        for (int c = ch * 64; c < ch * 64 + 64; ++c) {
            float v = Hs[r * HSTRIDE + c];
            s += v; s2 += v * v;
        }
        red[r * 16 + ch] = s;
        red2[r * 16 + ch] = s2;
    }
    __syncthreads();
    if (tid < 16) {
        float s = 0.0f, s2 = 0.0f;
        for (int ch = 0; ch < 16; ++ch) { s += red[tid * 16 + ch]; s2 += red2[tid * 16 + ch]; }
        float mu  = s * (1.0f / HID);
        float var = s2 * (1.0f / HID) - mu * mu;
        muS[tid]  = mu;
        invS[tid] = rsqrtf(var + EPSV);
    }
    __syncthreads();

    // Normalize + affine + leaky_relu -> f16
    for (int e = tid; e < 16 * HID; e += 256) {
        int r = e >> 10, c = e & (HID - 1);
        float v = (Hs[r * HSTRIDE + c] - muS[r]) * invS[r] * g2[c] + beta2[c];
        Hout[(size_t)(mBase + r) * HID + c] = (_Float16)lrelu(v);
    }
}

// ---------------------------------------------------------------------------
// GEMM2: H2[16 x 1024] @ W3[1024 x 256] + b3 -> f32 out
// A tile staged in LDS; waves cover 128 cols, 2 tiles per wave in registers.
// ---------------------------------------------------------------------------
__global__ void __launch_bounds__(256, 1)
k_gemm2(const _Float16* __restrict__ H2,
        const _Float16* __restrict__ W3T,
        const float* __restrict__ b3,
        float* __restrict__ out) {
    __shared__ _Float16 As[16 * ASTRIDE];                      // ~33 KB

    const int tid   = threadIdx.x;
    const int lane  = tid & 31;
    const int wave  = tid >> 5;
    const int mBase = blockIdx.x * 16;
    const int l15   = lane & 15;
    const int g     = lane >> 4;

    for (int c = tid; c < 16 * (HID / 8); c += 256) {
        int r  = c >> 7;
        int co = (c & 127) * 8;
        stage16B(H2 + (size_t)(mBase + r) * HID + co, As + r * ASTRIDE + co);
    }
    stage_join();
    __syncthreads();

    const _Float16* bPtr = W3T + (size_t)(wave * 16 + l15) * HID + g * 16;
    const _Float16* aLds = As + l15 * ASTRIDE + g * 16;

    v8f acc[2] = {};
    for (int kb = 0; kb < HID / 32; ++kb) {
        F16Frag af;
        af.h[0] = *(const v8h*)(aLds + kb * 32);
        af.h[1] = *(const v8h*)(aLds + kb * 32 + 8);
        F16Frag bf[2];
#pragma unroll
        for (int nc = 0; nc < 2; ++nc) {
            const _Float16* bp = bPtr + (size_t)nc * (128 * HID) + kb * 32;
            bf[nc].h[0] = *(const v8h*)(bp);
            bf[nc].h[1] = *(const v8h*)(bp + 8);
        }
        if (kb + 4 < HID / 32) {
#pragma unroll
            for (int nc = 0; nc < 2; ++nc)
                __builtin_prefetch(bPtr + (size_t)nc * (128 * HID) + (kb + 4) * 32, 0, 3);
        }
#pragma unroll
        for (int nc = 0; nc < 2; ++nc) {
            acc[nc] = __builtin_amdgcn_wmma_f32_16x16x32_f16(
                false, af.v, false, bf[nc].v, (short)0, acc[nc], false, false);
        }
    }
#pragma unroll
    for (int nc = 0; nc < 2; ++nc) {
        int col = nc * 128 + wave * 16 + l15;
        float bias = b3[col];
#pragma unroll
        for (int v = 0; v < 8; ++v) {
            int row = v + g * 8;
            out[(size_t)(mBase + row) * OUTD + col] = acc[nc][v] + bias;
        }
    }
}

// ---------------------------------------------------------------------------
extern "C" void kernel_launch(void* const* d_in, const int* in_sizes, int n_in,
                              void* d_out, int out_size, void* d_ws, size_t ws_size,
                              hipStream_t stream) {
    (void)in_sizes; (void)n_in; (void)out_size; (void)ws_size;
    const int*   x     = (const int*)d_in[0];
    const float* W1    = (const float*)d_in[1];
    const float* b1    = (const float*)d_in[2];
    const float* W2    = (const float*)d_in[3];
    const float* b2    = (const float*)d_in[4];
    const float* g2    = (const float*)d_in[5];
    const float* beta2 = (const float*)d_in[6];
    const float* W3    = (const float*)d_in[7];
    const float* b3    = (const float*)d_in[8];
    float* out = (float*)d_out;

    char* ws = (char*)d_ws;
    _Float16* A   = (_Float16*)(ws);                                 // 16 MB
    _Float16* H2  = (_Float16*)(ws + (size_t)16 * 1024 * 1024);      // 16 MB
    _Float16* W2T = (_Float16*)(ws + (size_t)32 * 1024 * 1024);      //  2 MB
    _Float16* W3T = (_Float16*)(ws + (size_t)34 * 1024 * 1024);      // 0.5 MB

    k_convW2<<<(HID * HID) / 256, 256, 0, stream>>>(W2, W2T);
    k_convW3<<<(OUTD * HID) / 256, 256, 0, stream>>>(W3, W3T);
    k_embed<<<BATCH, 256, 0, stream>>>(x, W1, b1, A);

    size_t smemB = (size_t)16 * ASTRIDE * 2                       // A tile (f16)
                 + (size_t)(16 * HSTRIDE + 256 + 256 + 16 + 16) * sizeof(float);
    k_gemm1_ln<<<BATCH / 16, 256, smemB, stream>>>(A, W2T, b2, g2, beta2, H2);
    k_gemm2<<<BATCH / 16, 256, 0, stream>>>(H2, W3T, b3, out);
}
